// Mambalayer_seq_20770461843911
// MI455X (gfx1250) — matrix operand order repaired and checked by reading, not compiled, not run on hardware
//
#include <hip/hip_runtime.h>
#include <hip/hip_bf16.h>

// ---------------- CDNA5 WMMA types ----------------
typedef __attribute__((ext_vector_type(16))) __bf16 v16bf;
typedef __attribute__((ext_vector_type(8)))  __bf16 v8bf;
typedef __attribute__((ext_vector_type(4)))  __bf16 v4bf;
typedef __attribute__((ext_vector_type(8)))  float  v8f;

#define D_MODEL 512
#define D_STATE 16
#define D_INNER 1024
#define DT_RANK 32
#define BATCH   2
#define SEQLEN  2048
#define MROWS   (BATCH * SEQLEN)   // 4096

__device__ __forceinline__ float silu_f(float v) {
    return v / (1.0f + __expf(-v));
}

// ---------------- fp32 -> bf16 convert (4 elems / thread) ----------------
__global__ void cvt_f32_bf16(const float* __restrict__ in, __bf16* __restrict__ out, int n) {
    int idx = (blockIdx.x * blockDim.x + threadIdx.x) * 4;
    if (idx >= n) return;
    float4 v = *(const float4*)(in + idx);
    v4bf o;
    o[0] = (__bf16)v.x; o[1] = (__bf16)v.y; o[2] = (__bf16)v.z; o[3] = (__bf16)v.w;
    *(v4bf*)(out + idx) = o;
}

// ---------------- bf16 WMMA GEMM: C[M,N] = A[M,K] * B[N,K]^T ----------------
// Block = 256 threads (8 waves, wave32) -> 128(M) x 64(N) macro-tile.
//   * B tile (64 rows x 32 K halves = 4 KB) is staged into LDS once per K-step
//     with gfx1250 async copies (global_load_async_to_lds_b128, ASYNCcnt).
//   * Each wave owns a 16-row strip and 4 accumulators (16x64): one A fragment
//     feeds 4 v_wmma_f32_16x16x32_bf16 per K-step.
// Fragment layouts used here (contiguous k order, same for A and B):
//   A 16x32: lanes 0-15 rows M0..15 K0..15; lanes 16-31 K16..31
//   B 32x16: lanes 0-15 K0..15 of col N=lane; lanes 16-31 K16..31 of col N=lane-16
__global__ void gemm_bf16_wmma(const __bf16* __restrict__ A,
                               const __bf16* __restrict__ B,
                               float* __restrict__ C,
                               int M, int N, int K) {
    __shared__ __align__(32) __bf16 sB[64 * 32];       // 4 KB B tile

    const int lane = threadIdx.x & 31;
    const int wave = threadIdx.x >> 5;
    const int n0   = blockIdx.x * 64;
    const int m0   = blockIdx.y * 128 + wave * 16;

    const int r  = lane & 15;
    const bool lo = (lane < 16);
    const int kloA = lo ? 0 : 16;                      // A fragment k base (halves)
    const int kloB = lo ? 0 : 16;                      // B fragment k base (halves)

    // cooperative B-tile staging assignment: 4 threads x 16B per 64B row
    const int brow = threadIdx.x >> 2;                 // 0..63
    const int bseg = threadIdx.x & 3;                  // 0..3
    const __bf16* gBrow = B + (size_t)(n0 + brow) * K + bseg * 8;
    const uint32_t ldsB =
        (uint32_t)(uintptr_t)(&sB[0]) + (uint32_t)(brow * 64 + bseg * 16);

    const __bf16* Arow = A + (size_t)(m0 + r) * K;

    v8f acc0 = {}, acc1 = {}, acc2 = {}, acc3 = {};

    for (int kt = 0; kt < K; kt += 32) {
        // ---- async copy B tile to LDS (tracked by ASYNCcnt) ----
        {
            uint64_t g = (uint64_t)(uintptr_t)(gBrow + kt);
            asm volatile("global_load_async_to_lds_b128 %0, %1, off"
                         :: "v"(ldsB), "v"(g) : "memory");
            asm volatile("s_wait_asynccnt 0x0" ::: "memory");
        }
        __syncthreads();

        // ---- A fragment straight from global ----
        if (kt + 32 < K)
            __builtin_prefetch(Arow + kt + 32 + kloA, 0, 3);
        v8bf a_lo = *(const v8bf*)(Arow + kt + kloA);
        v8bf a_hi = *(const v8bf*)(Arow + kt + kloA + 8);
        v16bf a = __builtin_shufflevector(a_lo, a_hi,
                    0,1,2,3,4,5,6,7,8,9,10,11,12,13,14,15);

        // ---- 4 B fragments from LDS, 4 WMMAs reusing A ----
        v16bf b0 = *(const v16bf*)(sB + (0 * 16 + r) * 32 + kloB);
        v16bf b1 = *(const v16bf*)(sB + (1 * 16 + r) * 32 + kloB);
        v16bf b2 = *(const v16bf*)(sB + (2 * 16 + r) * 32 + kloB);
        v16bf b3 = *(const v16bf*)(sB + (3 * 16 + r) * 32 + kloB);
        acc0 = __builtin_amdgcn_wmma_f32_16x16x32_bf16(false, a, false, b0,
                                                       (short)0, acc0, false, false);
        acc1 = __builtin_amdgcn_wmma_f32_16x16x32_bf16(false, a, false, b1,
                                                       (short)0, acc1, false, false);
        acc2 = __builtin_amdgcn_wmma_f32_16x16x32_bf16(false, a, false, b2,
                                                       (short)0, acc2, false, false);
        acc3 = __builtin_amdgcn_wmma_f32_16x16x32_bf16(false, a, false, b3,
                                                       (short)0, acc3, false, false);
        __syncthreads();                               // protect sB reuse
    }

    // C/D layout: VGPR i -> row m0 + i + (lane<16 ? 0 : 8), col = n-base + (lane&15)
    float* Cp = C + (size_t)(m0 + (lo ? 0 : 8)) * N + n0 + r;
    #pragma unroll
    for (int i = 0; i < 8; i++) {
        Cp[(size_t)i * N +  0] = acc0[i];
        Cp[(size_t)i * N + 16] = acc1[i];
        Cp[(size_t)i * N + 32] = acc2[i];
        Cp[(size_t)i * N + 48] = acc3[i];
    }
}

// ---------------- causal depthwise conv1d (k=4) + bias + SiLU ----------------
__global__ void conv_silu(const float* __restrict__ xz,
                          const float* __restrict__ conv_w,
                          const float* __restrict__ conv_b,
                          float* __restrict__ u_f, __bf16* __restrict__ u_b) {
    int idx = blockIdx.x * blockDim.x + threadIdx.x;   // B*L*D_INNER threads
    int d  = idx & (D_INNER - 1);
    int bl = idx >> 10;
    int l  = bl & (SEQLEN - 1);
    int b  = bl >> 11;
    float acc = conv_b[d];
    #pragma unroll
    for (int j = 0; j < 4; j++) {
        int t = l - 3 + j;
        if (t >= 0)
            acc += conv_w[d * 4 + j] * xz[((size_t)(b * SEQLEN + t)) * (2 * D_INNER) + d];
    }
    float s = silu_f(acc);
    u_f[idx] = s;
    u_b[idx] = (__bf16)s;
}

// ---------------- delta = softplus(dt @ dt_proj_w^T + b) ----------------
__global__ void dt_delta(const float* __restrict__ x_dbl,
                         const float* __restrict__ dtw,
                         const float* __restrict__ dtb,
                         float* __restrict__ delta) {
    int idx = blockIdx.x * blockDim.x + threadIdx.x;   // B*L*D_INNER
    int d  = idx & (D_INNER - 1);
    int bl = idx >> 10;
    const float* dt = x_dbl + (size_t)bl * (DT_RANK + 2 * D_STATE);
    float acc = dtb[d];
    #pragma unroll
    for (int rr = 0; rr < DT_RANK; rr++) acc += dt[rr] * dtw[d * DT_RANK + rr];
    delta[idx] = (acc > 20.0f) ? acc : log1pf(__expf(acc));
}

// ---------------- selective scan ----------------
// One wave handles 2 (b,d) channels: lanes 0-15 = states of channel 2g,
// lanes 16-31 = states of channel 2g+1. Sequential over L; shuffle-reduce 16 states.
__global__ void selective_scan(const float* __restrict__ delta,
                               const float* __restrict__ u_f,
                               const float* __restrict__ x_dbl,
                               const float* __restrict__ A_log,
                               const float* __restrict__ Dp,
                               float* __restrict__ y) {
    int lane = threadIdx.x & 31;
    int gid  = (blockIdx.x * blockDim.x + threadIdx.x) >> 5;
    int half = lane >> 4;
    int n    = lane & 15;
    int ch   = gid * 2 + half;                 // 0 .. 2047
    int b    = ch >> 10;
    int d    = ch & (D_INNER - 1);

    const float An = -__expf(A_log[d * D_STATE + n]);
    const float Dd = Dp[d];
    float h = 0.0f;

    const size_t rowDU = (size_t)b * SEQLEN * D_INNER + d;
    const size_t rowBC = (size_t)b * SEQLEN * (DT_RANK + 2 * D_STATE);

    for (int t = 0; t < SEQLEN; t++) {
        float dt_t = delta[rowDU + (size_t)t * D_INNER];
        float u_t  = u_f [rowDU + (size_t)t * D_INNER];
        const float* bc = x_dbl + rowBC + (size_t)t * (DT_RANK + 2 * D_STATE);
        float Bn = bc[DT_RANK + n];
        float Cn = bc[DT_RANK + D_STATE + n];
        float dA = __expf(dt_t * An);
        h = dA * h + dt_t * u_t * Bn;
        float p = h * Cn;
        p += __shfl_xor(p, 1, 32);
        p += __shfl_xor(p, 2, 32);
        p += __shfl_xor(p, 4, 32);
        p += __shfl_xor(p, 8, 32);
        if (n == 0)
            y[rowDU + (size_t)t * D_INNER] = p + u_t * Dd;
    }
}

// ---------------- gate: yg = y * silu(z) -> bf16 ----------------
__global__ void gate_silu(const float* __restrict__ xz,
                          const float* __restrict__ y,
                          __bf16* __restrict__ ygb) {
    int idx = blockIdx.x * blockDim.x + threadIdx.x;   // B*L*D_INNER
    int d  = idx & (D_INNER - 1);
    int bl = idx >> 10;
    float z = xz[(size_t)bl * (2 * D_INNER) + D_INNER + d];
    ygb[idx] = (__bf16)(y[idx] * silu_f(z));
}

// ---------------- host-side orchestration ----------------
extern "C" void kernel_launch(void* const* d_in, const int* in_sizes, int n_in,
                              void* d_out, int out_size, void* d_ws, size_t ws_size,
                              hipStream_t stream) {
    const float* x         = (const float*)d_in[0];   // (2,2048,512)
    const float* in_proj_w = (const float*)d_in[1];   // (2048,512)
    const float* conv_w    = (const float*)d_in[2];   // (1024,1,4)
    const float* conv_b    = (const float*)d_in[3];   // (1024,)
    const float* x_proj_w  = (const float*)d_in[4];   // (64,1024)
    const float* dt_proj_w = (const float*)d_in[5];   // (1024,32)
    const float* dt_proj_b = (const float*)d_in[6];   // (1024,)
    const float* A_log     = (const float*)d_in[7];   // (1024,16)
    const float* Dp        = (const float*)d_in[8];   // (1024,)
    const float* out_proj_w= (const float*)d_in[9];   // (512,1024)
    float* out = (float*)d_out;                       // (2,2048,512)

    // workspace layout (all sizes multiples of 256B)
    char* ws = (char*)d_ws;
    size_t off = 0;
    float*  xz_f     = (float*) (ws + off); off += (size_t)MROWS * 2 * D_INNER * 4; // 32 MB
    __bf16* x_b      = (__bf16*)(ws + off); off += (size_t)MROWS * D_MODEL * 2;     // 4 MB
    __bf16* w_in_b   = (__bf16*)(ws + off); off += (size_t)2 * D_INNER * D_MODEL * 2;
    float*  u_f      = (float*) (ws + off); off += (size_t)MROWS * D_INNER * 4;
    __bf16* u_b      = (__bf16*)(ws + off); off += (size_t)MROWS * D_INNER * 2;
    __bf16* w_xp_b   = (__bf16*)(ws + off); off += (size_t)(DT_RANK + 2*D_STATE) * D_INNER * 2;
    float*  x_dbl_f  = (float*) (ws + off); off += (size_t)MROWS * (DT_RANK + 2*D_STATE) * 4;
    float*  delta_f  = (float*) (ws + off); off += (size_t)MROWS * D_INNER * 4;
    float*  y_f      = (float*) (ws + off); off += (size_t)MROWS * D_INNER * 4;
    __bf16* yg_b     = (__bf16*)(ws + off); off += (size_t)MROWS * D_INNER * 2;
    __bf16* w_out_b  = (__bf16*)(ws + off); off += (size_t)D_MODEL * D_INNER * 2;
    (void)ws_size; (void)off; (void)n_in; (void)in_sizes; (void)out_size;

    // 1) convert activations & weights to bf16
    {
        int n;
        n = MROWS * D_MODEL;
        cvt_f32_bf16<<<dim3(n/4/256), dim3(256), 0, stream>>>(x, x_b, n);
        n = 2 * D_INNER * D_MODEL;
        cvt_f32_bf16<<<dim3(n/4/256), dim3(256), 0, stream>>>(in_proj_w, w_in_b, n);
        n = (DT_RANK + 2*D_STATE) * D_INNER;
        cvt_f32_bf16<<<dim3(n/4/256), dim3(256), 0, stream>>>(x_proj_w, w_xp_b, n);
        n = D_MODEL * D_INNER;
        cvt_f32_bf16<<<dim3(n/4/256), dim3(256), 0, stream>>>(out_proj_w, w_out_b, n);
    }

    // 2) in_proj: xz = x @ in_proj_w^T   (M=4096, K=512, N=2048)
    gemm_bf16_wmma<<<dim3((2*D_INNER)/64, MROWS/128), dim3(256), 0, stream>>>(
        x_b, w_in_b, xz_f, MROWS, 2*D_INNER, D_MODEL);

    // 3) depthwise conv + SiLU
    {
        int n = MROWS * D_INNER;                       // 4M
        conv_silu<<<dim3(n/256), dim3(256), 0, stream>>>(xz_f, conv_w, conv_b, u_f, u_b);
    }

    // 4) x_proj: x_dbl = u @ x_proj_w^T  (M=4096, K=1024, N=64)
    gemm_bf16_wmma<<<dim3((DT_RANK + 2*D_STATE)/64, MROWS/128), dim3(256), 0, stream>>>(
        u_b, w_xp_b, x_dbl_f, MROWS, DT_RANK + 2*D_STATE, D_INNER);

    // 5) delta = softplus(dt @ dt_proj_w^T + b)   (K=32, scalar fp32)
    {
        int n = MROWS * D_INNER;
        dt_delta<<<dim3(n/256), dim3(256), 0, stream>>>(x_dbl_f, dt_proj_w, dt_proj_b, delta_f);
    }

    // 6) selective scan: 2048 channels, 2 per wave -> 1024 waves
    {
        int threads = (BATCH * D_INNER / 2) * 32;      // 32768
        selective_scan<<<dim3(threads/256), dim3(256), 0, stream>>>(
            delta_f, u_f, x_dbl_f, A_log, Dp, y_f);
    }

    // 7) gate
    {
        int n = MROWS * D_INNER;
        gate_silu<<<dim3(n/256), dim3(256), 0, stream>>>(xz_f, y_f, yg_b);
    }

    // 8) out_proj: out = yg @ out_proj_w^T  (M=4096, K=1024, N=512)
    gemm_bf16_wmma<<<dim3(D_MODEL/64, MROWS/128), dim3(256), 0, stream>>>(
        yg_b, w_out_b, out, MROWS, D_MODEL, D_INNER);
}
